// _fasterRCNN_51754355917516
// MI455X (gfx1250) — matrix operand order repaired and checked by reading, compiled, not run on hardware
//
#include <hip/hip_runtime.h>
#include <hip/hip_bf16.h>
#include <stdint.h>

#define NBOX 2048
#define DIM  4096
#define NCLS 21
#define OUTC 105

typedef __attribute__((ext_vector_type(16))) __bf16 v16bf;
typedef __attribute__((ext_vector_type(8)))  float  v8f;
typedef __attribute__((ext_vector_type(4)))  int    v4i;

__device__ __forceinline__ unsigned short f32_to_bf16(float f) {
  unsigned int u = __float_as_uint(f);
  unsigned int r = (u + 0x7FFFu + ((u >> 16) & 1u)) >> 16;  // RNE
  return (unsigned short)r;
}

// Hash-based priority standing in for jax.random.uniform draws.
__device__ __forceinline__ float hpri(unsigned int i, unsigned int j, unsigned int salt) {
  unsigned int x = i * 2048u + j + salt * 0x9E3779B9u;
  x ^= x >> 16; x *= 0x7feb352dU; x ^= x >> 15; x *= 0x846ca68bU; x ^= x >> 16;
  return (float)(x >> 8) * (1.0f / 16777216.0f);
}

template <int KK>
__device__ __forceinline__ void insert_topk(float (&p)[KK], int (&id)[KK], float pri, int j) {
  if (pri <= p[KK - 1]) return;
  p[KK - 1] = pri; id[KK - 1] = j;
  #pragma unroll
  for (int t = KK - 1; t > 0; --t) {
    if (p[t] > p[t - 1]) {
      float tp = p[t]; p[t] = p[t - 1]; p[t - 1] = tp;
      int  ti = id[t]; id[t] = id[t - 1]; id[t - 1] = ti;
    }
  }
}

// -------- CDNA5 async global->LDS copy (ASYNCcnt path), guarded for toolchain
__device__ __forceinline__ void async_cp16(const unsigned short* g, unsigned short* l) {
#if __has_builtin(__builtin_amdgcn_global_load_async_to_lds_b128)
  __builtin_amdgcn_global_load_async_to_lds_b128(
      (__attribute__((address_space(1))) v4i*)(g),
      (__attribute__((address_space(3))) v4i*)(l), 0, 0);
#else
  *(uint4*)l = *(const uint4*)g;  // synchronous fallback
#endif
}

template <int N>
__device__ __forceinline__ void wait_async() {
#if __has_builtin(__builtin_amdgcn_s_wait_asynccnt)
  __builtin_amdgcn_s_wait_asynccnt(N);
#else
  asm volatile("s_wait_asynccnt %0" ::"i"(N) : "memory");
#endif
}

// ---------------------------------------------------------------- converts
__global__ void k_conv_bf16(const float* __restrict__ in, unsigned short* __restrict__ out, size_t n) {
  size_t i = (size_t)blockIdx.x * blockDim.x + threadIdx.x;
  size_t stride = (size_t)gridDim.x * blockDim.x;
  for (; i < n; i += stride) out[i] = f32_to_bf16(in[i]);
}

// in: K x N fp32 (row-major); out: N x K bf16 (transposed) — LDS-tiled for coalescing
__global__ void k_trans_bf16(const float* __restrict__ in, unsigned short* __restrict__ out, int K, int N) {
  __shared__ unsigned short tile[32][33];
  int k0 = blockIdx.y * 32, n0 = blockIdx.x * 32;
  for (int r = threadIdx.y; r < 32; r += 8)
    tile[r][threadIdx.x] = f32_to_bf16(in[(size_t)(k0 + r) * N + n0 + threadIdx.x]);
  __syncthreads();
  for (int r = threadIdx.y; r < 32; r += 8)
    out[(size_t)(n0 + r) * K + k0 + threadIdx.x] = tile[threadIdx.x][r];
}

// ---------------------------------------------------------------- row norms
__global__ void k_invlen(const float* __restrict__ F, float* __restrict__ invlen) {
  int w = (blockIdx.x * blockDim.x + threadIdx.x) >> 5;
  int lane = threadIdx.x & 31;
  if (w >= NBOX) return;
  const float* row = F + (size_t)w * DIM;
  float s = 0.f;
  for (int k = lane; k < DIM; k += 32) { float v = row[k]; s += v * v; }
  #pragma unroll
  for (int off = 16; off; off >>= 1) s += __shfl_xor(s, off, 32);
  if (lane == 0) invlen[w] = rsqrtf(s);
}

// ---------------------------------------------------------------- mask builder (sparse children)
__global__ void k_mask(const float* __restrict__ rois, int* __restrict__ child) {
  int i = blockIdx.x * blockDim.x + threadIdx.x;
  if (i >= NBOX) return;
  float x1i = rois[i * 5 + 1], y1i = rois[i * 5 + 2];
  float x2i = rois[i * 5 + 3], y2i = rois[i * 5 + 4];
  float areai = (x2i - x1i + 1.f) * (y2i - y1i + 1.f);
  float cxi = (x1i + x2i) * 0.5f, cyi = (y1i + y2i) * 0.5f;

  float pov[6]; int iov[6]; float pds[6]; int ids[6];
  #pragma unroll
  for (int t = 0; t < 6; ++t) { pov[t] = -1.f; iov[t] = -1; pds[t] = -1.f; ids[t] = -1; }

  for (int j = 0; j < NBOX; ++j) {
    float x1j = rois[j * 5 + 1], y1j = rois[j * 5 + 2];
    float x2j = rois[j * 5 + 3], y2j = rois[j * 5 + 4];
    float areaj = (x2j - x1j + 1.f) * (y2j - y1j + 1.f);
    float iw = fminf(x2i, x2j) - fmaxf(x1i, x1j) + 1.f; iw = fmaxf(iw, 0.f);
    float ih = fminf(y2i, y2j) - fmaxf(y1i, y1j) + 1.f; ih = fmaxf(ih, 0.f);
    float inter = iw * ih;
    float iou = inter / (areai + areaj - inter);
    float dx = cxi - (x1j + x2j) * 0.5f, dy = cyi - (y1j + y2j) * 0.5f;
    float dist = sqrtf(dx * dx + dy * dy) * (1.f / 1000.f);
    if (iou >= 0.7f)   insert_topk<6>(pov, iov, hpri(i, j, 1u), j);
    if (dist < 0.01f)  insert_topk<6>(pds, ids, hpri(i, j, 2u), j);
  }
  // union (dedupe) of the two top-6 lists
  int u[12]; int un = 0;
  #pragma unroll
  for (int t = 0; t < 12; ++t) {
    int j = (t < 6) ? iov[t] : ids[t - 6];
    if (j < 0) continue;
    bool dup = false;
    for (int q = 0; q < un; ++q) if (u[q] == j) dup = true;
    if (!dup) u[un++] = j;
  }
  // top-4 children by a third priority draw
  float p4[4]; int c4[4];
  #pragma unroll
  for (int t = 0; t < 4; ++t) { p4[t] = -1.f; c4[t] = -1; }
  for (int t = 0; t < un; ++t) insert_topk<4>(p4, c4, hpri(i, u[t], 3u), u[t]);
  #pragma unroll
  for (int t = 0; t < 4; ++t) child[i * 4 + t] = c4[t];
}

// ---------------------------------------------------------------- sparse cosine sim values
__global__ void k_simpair(const float* __restrict__ F, const int* __restrict__ child,
                          const float* __restrict__ invlen, float* __restrict__ adjv) {
  int i = blockIdx.x;
  int w = threadIdx.x >> 5, lane = threadIdx.x & 31;
  int j = child[i * 4 + w];
  float val = 0.f;
  if (j >= 0 && j != i) {  // j == i is covered by the unit diagonal term
    const float* a = F + (size_t)i * DIM;
    const float* b = F + (size_t)j * DIM;
    float s = 0.f;
    for (int k = lane; k < DIM; k += 32) s += a[k] * b[k];
    #pragma unroll
    for (int off = 16; off; off >>= 1) s += __shfl_xor(s, off, 32);
    val = s * invlen[i] * invlen[j];
  }
  if (lane == 0) adjv[i * 4 + w] = val;
}

// ---------------------------------------------------------------- bf16 WMMA GEMM: C[MxN] = A[MxK] * Bt[NxK]^T
// Block = 8 waves; block tile 128x64; K staged through double-buffered LDS via
// async global->LDS b128 copies (ASYNCcnt). Rows padded to 40 shorts (80B) to
// break LDS bank conflicts on the ds_load_b128 fragment reads.
__global__ void __launch_bounds__(256)
k_gemm_bf16(const unsigned short* __restrict__ A,
            const unsigned short* __restrict__ Bt,
            float* __restrict__ C, int M, int N, int K) {
  __shared__ __align__(16) unsigned short lA[2][128][40];  // 20.0 KB
  __shared__ __align__(16) unsigned short lB[2][64][40];   // 10.0 KB

  const int tid  = threadIdx.x;
  const int lane = tid & 31;
  const int wave = tid >> 5;
  const int hl   = lane >> 4;   // half-wave select
  const int l16  = lane & 15;
  const int rowB = blockIdx.y * 128;
  const int colB = blockIdx.x * 64;

  // copy mapping: 4 lanes cover one contiguous 64B (32-short) row slice
  const int arow = tid >> 2;          // 0..63
  const int seg  = (tid & 3) * 8;     // short offset {0,8,16,24}

  const unsigned short* gA0 = A  + (size_t)(rowB + arow)      * K + seg;
  const unsigned short* gA1 = A  + (size_t)(rowB + arow + 64) * K + seg;
  const unsigned short* gB  = Bt + (size_t)(colB + arow)      * K + seg;

  auto issue = [&](int k0, int buf) {
    async_cp16(gA0 + k0, &lA[buf][arow][seg]);
    async_cp16(gA1 + k0, &lA[buf][arow + 64][seg]);
    async_cp16(gB  + k0, &lB[buf][arow][seg]);
  };

  v8f c0 = {}, c1 = {}, c2 = {}, c3 = {};
  union U { uint4 u[2]; v16bf v; };

  const int KT = K >> 5;  // K/32 k-steps
  issue(0, 0);
  for (int kt = 0; kt < KT; ++kt) {
    const int buf = kt & 1;
    if (kt + 1 < KT) { issue((kt + 1) << 5, buf ^ 1); wait_async<3>(); }
    else             { wait_async<0>(); }
    __syncthreads();

    // A fragment: lane = row l16; 16 contiguous K from hl*16
    const unsigned short* ar = &lA[buf][wave * 16 + l16][0];
    U a, b0, b1, b2, b3;
    a.u[0] = *(const uint4*)(ar + hl * 16);
    a.u[1] = *(const uint4*)(ar + hl * 16 + 8);
    // B fragment: lane = col l16; 16 contiguous K from hl*16
    const unsigned short* br0 = &lB[buf][ 0 + l16][hl * 16];
    const unsigned short* br1 = &lB[buf][16 + l16][hl * 16];
    const unsigned short* br2 = &lB[buf][32 + l16][hl * 16];
    const unsigned short* br3 = &lB[buf][48 + l16][hl * 16];
    b0.u[0] = *(const uint4*)(br0); b0.u[1] = *(const uint4*)(br0 + 8);
    b1.u[0] = *(const uint4*)(br1); b1.u[1] = *(const uint4*)(br1 + 8);
    b2.u[0] = *(const uint4*)(br2); b2.u[1] = *(const uint4*)(br2 + 8);
    b3.u[0] = *(const uint4*)(br3); b3.u[1] = *(const uint4*)(br3 + 8);

    c0 = __builtin_amdgcn_wmma_f32_16x16x32_bf16(false, a.v, false, b0.v, (short)0, c0, false, false);
    c1 = __builtin_amdgcn_wmma_f32_16x16x32_bf16(false, a.v, false, b1.v, (short)0, c1, false, false);
    c2 = __builtin_amdgcn_wmma_f32_16x16x32_bf16(false, a.v, false, b2.v, (short)0, c2, false, false);
    c3 = __builtin_amdgcn_wmma_f32_16x16x32_bf16(false, a.v, false, b3.v, (short)0, c3, false, false);

    __syncthreads();  // protect buffer reuse two iterations out
  }

  // C/D layout: VGPR r -> row r (lanes 0-15) or r+8 (lanes 16-31); col = l16
  const int mbase = rowB + wave * 16 + hl * 8;
  #pragma unroll
  for (int r = 0; r < 8; ++r) {
    float* crow = C + (size_t)(mbase + r) * N + colB + l16;
    crow[ 0] = c0[r];
    crow[16] = c1[r];
    crow[32] = c2[r];
    crow[48] = c3[r];
  }
}

// ---------------------------------------------------------------- sparse aggregation: relu(adj @ H + b)
template <bool BF16OUT>
__global__ void k_agg(const float* __restrict__ H, const float* __restrict__ bias,
                      const int* __restrict__ child, const float* __restrict__ adjv,
                      void* __restrict__ out) {
  int i = blockIdx.x;
  int cj[4]; float cv[4];
  #pragma unroll
  for (int t = 0; t < 4; ++t) { cj[t] = child[i * 4 + t]; cv[t] = adjv[i * 4 + t]; }
  for (int k = threadIdx.x; k < DIM; k += blockDim.x) {
    float v = H[(size_t)i * DIM + k];  // diagonal coefficient == 1
    #pragma unroll
    for (int t = 0; t < 4; ++t)
      if (cj[t] >= 0 && cv[t] != 0.f) v += cv[t] * H[(size_t)cj[t] * DIM + k];
    v = fmaxf(v + bias[k], 0.f);
    if (BF16OUT) ((unsigned short*)out)[(size_t)i * DIM + k] = f32_to_bf16(v);
    else         ((float*)out)[(size_t)i * DIM + k] = v;
  }
}

// ---------------------------------------------------------------- head: softmax(x@Wc+bc) || x@Wb+bb
__global__ void k_head(const float* __restrict__ X,
                       const float* __restrict__ Wc, const float* __restrict__ bc,
                       const float* __restrict__ Wb, const float* __restrict__ bb,
                       float* __restrict__ out) {
  __shared__ float logits[NCLS];
  int i = blockIdx.x, c = threadIdx.x;
  const float* x = X + (size_t)i * DIM;
  float acc = 0.f;
  if (c < OUTC) {
    if (c < NCLS) {
      for (int k = 0; k < DIM; ++k) acc += x[k] * Wc[(size_t)k * NCLS + c];
      acc += bc[c];
    } else {
      int cc = c - NCLS;
      for (int k = 0; k < DIM; ++k) acc += x[k] * Wb[(size_t)k * (4 * NCLS) + cc];
      acc += bb[cc];
    }
  }
  if (c < NCLS) logits[c] = acc;
  __syncthreads();
  if (c < OUTC) {
    if (c < NCLS) {
      float m = -1e30f;
      for (int t = 0; t < NCLS; ++t) m = fmaxf(m, logits[t]);
      float s = 0.f;
      for (int t = 0; t < NCLS; ++t) s += expf(logits[t] - m);
      out[(size_t)i * OUTC + c] = expf(acc - m) / s;
    } else {
      out[(size_t)i * OUTC + c] = acc;
    }
  }
}

// ----------------------------------------------------------------
extern "C" void kernel_launch(void* const* d_in, const int* in_sizes, int n_in,
                              void* d_out, int out_size, void* d_ws, size_t ws_size,
                              hipStream_t stream) {
  (void)in_sizes; (void)n_in; (void)out_size; (void)ws_size;
  const float* rois   = (const float*)d_in[0];
  const float* pooled = (const float*)d_in[1];
  const float* W1     = (const float*)d_in[2];
  const float* b1     = (const float*)d_in[3];
  const float* W2     = (const float*)d_in[4];
  const float* b2     = (const float*)d_in[5];
  const float* Wc     = (const float*)d_in[6];
  const float* bc     = (const float*)d_in[7];
  const float* Wb     = (const float*)d_in[8];
  const float* bb     = (const float*)d_in[9];
  float* out = (float*)d_out;

  size_t off = 0;
  auto take = [&](size_t bytes) -> void* {
    void* p = (char*)d_ws + off;
    off += (bytes + 255) & ~(size_t)255;
    return p;
  };
  unsigned short* bfA = (unsigned short*)take((size_t)NBOX * DIM * 2);
  unsigned short* bfW = (unsigned short*)take((size_t)DIM * DIM * 2);   // W1t then W2t
  unsigned short* bfX = (unsigned short*)take((size_t)NBOX * DIM * 2);
  float* H      = (float*)take((size_t)NBOX * DIM * 4);                  // H1 then H2
  float* X2     = (float*)take((size_t)NBOX * DIM * 4);
  float* invlen = (float*)take((size_t)NBOX * 4);
  int*   child  = (int*)take((size_t)NBOX * 4 * sizeof(int));
  float* adjv   = (float*)take((size_t)NBOX * 4 * sizeof(float));

  // geometry / sparsity path
  k_invlen<<<NBOX / 8, 256, 0, stream>>>(pooled, invlen);
  k_mask<<<NBOX / 256, 256, 0, stream>>>(rois, child);
  k_simpair<<<NBOX, 128, 0, stream>>>(pooled, child, invlen, adjv);

  // precision convert
  k_conv_bf16<<<8192, 256, 0, stream>>>(pooled, bfA, (size_t)NBOX * DIM);
  k_trans_bf16<<<dim3(DIM / 32, DIM / 32), dim3(32, 8), 0, stream>>>(W1, bfW, DIM, DIM);

  // layer 1: H1 = pooled @ W1 ; X1 = relu(adj@H1 + b1) -> bf16
  k_gemm_bf16<<<dim3(DIM / 64, NBOX / 128), 256, 0, stream>>>(bfA, bfW, H, NBOX, DIM, DIM);
  k_agg<true><<<NBOX, 256, 0, stream>>>(H, b1, child, adjv, bfX);

  // layer 2: H2 = X1 @ W2 ; X2 = relu(adj@H2 + b2) -> fp32
  k_trans_bf16<<<dim3(DIM / 32, DIM / 32), dim3(32, 8), 0, stream>>>(W2, bfW, DIM, DIM);
  k_gemm_bf16<<<dim3(DIM / 64, NBOX / 128), 256, 0, stream>>>(bfX, bfW, H, NBOX, DIM, DIM);
  k_agg<false><<<NBOX, 256, 0, stream>>>(H, b2, child, adjv, X2);

  // classification / bbox head
  k_head<<<NBOX, 128, 0, stream>>>(X2, Wc, bc, Wb, bb, out);
}
